// MovingAverageVectorQuantizer_83923660964604
// MI455X (gfx1250) — matrix operand order repaired and checked by reading, not compiled, not run on hardware
//
#include <hip/hip_runtime.h>

#define N_E   1024
#define EDIM  256
#define DECAYF 0.99f
#define EPSF   1e-5f
#define NVEC  131072            // 32*64*64 vectors
#define HW    4096              // 64*64

typedef __bf16  bf16_t;
typedef bf16_t  v16bf __attribute__((ext_vector_type(16)));
typedef bf16_t  v8bf  __attribute__((ext_vector_type(8)));
typedef bf16_t  v4bf  __attribute__((ext_vector_type(4)));
typedef float   v8f   __attribute__((ext_vector_type(8)));

// raw LDS byte offset (addrspace(3) ptrtoint) for async-to-LDS VDST operand
__device__ __forceinline__ unsigned lds_offset_of(const void* p) {
    return (unsigned)(unsigned long long)
           (__attribute__((address_space(3))) const char*)p;
}

// ---------------- workspace zero ----------------
__global__ void vq_zero(float* __restrict__ p, int n) {
    int i = blockIdx.x * blockDim.x + threadIdx.x;
    if (i < n) p[i] = 0.0f;
}

// ---------------- one-time fp32 -> bf16 codebook image ----------------
__global__ void vq_cvt(const float* __restrict__ emb, bf16_t* __restrict__ ebf) {
    int i = blockIdx.x * blockDim.x + threadIdx.x;   // float4 index, 65536 total
    float4 v = ((const float4*)emb)[i];
    v4bf pk;
    pk[0] = (bf16_t)v.x; pk[1] = (bf16_t)v.y;
    pk[2] = (bf16_t)v.z; pk[3] = (bf16_t)v.w;
    ((v4bf*)ebf)[i] = pk;
}

// ---------------- per-code squared norms ----------------
__global__ void vq_e2(const float* __restrict__ emb, float* __restrict__ e2) {
    int c = blockIdx.x * blockDim.x + threadIdx.x;
    if (c < N_E) {
        const float4* p = (const float4*)(emb + c * EDIM);
        float s = 0.0f;
        #pragma unroll 4
        for (int i = 0; i < EDIM / 4; ++i) {
            float4 v = p[i];
            s += v.x * v.x + v.y * v.y + v.z * v.z + v.w * v.w;
        }
        e2[c] = s;
    }
}

// ---------------- main fused kernel ----------------
// grid: 1024 blocks x 256 threads. Block handles 128 consecutive z-vectors.
// Wave w handles 16 vectors; WMMA tiles: M = 16 codes, N = 16 vectors, K = 32.
__global__ __launch_bounds__(256)
void vq_main(const float* __restrict__ z, const float* __restrict__ emb,
             const bf16_t* __restrict__ ebf, const float* __restrict__ e2,
             float* __restrict__ oZq, float* __restrict__ oIdx,
             float* __restrict__ wsCounts, float* __restrict__ wsEmbsum,
             float* __restrict__ wsLoss)
{
    __shared__ bf16_t sE[128 * EDIM];   // 64 KB: 128 codes x 256 dims, bf16
    __shared__ float  sE2[128];

    const int tid  = threadIdx.x;
    const int lane = tid & 31;
    const int wave = tid >> 5;
    const int half = lane >> 4;         // which 16-lane half
    const int col  = lane & 15;         // z-vector column within wave tile

    const int nBase = blockIdx.x * 128 + wave * 16;
    const int n     = nBase + col;
    const int img   = n >> 12;          // batch index (4096 = 64*64)
    const int hw    = n & (HW - 1);
    const int base  = img * EDIM * HW + hw;   // &z[img][0][hw]

    const unsigned sE_off = lds_offset_of(&sE[0]);

    // ---- preload the wave's 16 z-vectors as bf16 B-fragments (K = 256) ----
    // B(32x16): lanes 0-15 hold K 0-7 & 16-23, lanes 16-31 hold K 8-15 & 24-31 (2/VGPR)
    v16bf B[8];
    #pragma unroll
    for (int f = 0; f < 8; ++f) {
        v16bf bv;
        #pragma unroll
        for (int e = 0; e < 16; ++e) {
            int k = f * 32 + half * 8 + e + (e >= 8 ? 8 : 0);
            bv[e] = (bf16_t)z[base + k * HW];
        }
        B[f] = bv;
    }

    float minv = __builtin_inff();
    int   mini = 0;

    for (int ch = 0; ch < 8; ++ch) {
        __syncthreads();
        // ---- async DMA: 64 KB bf16 chunk (128 codes) global -> LDS ----
        const char* srcBf = (const char*)ebf + ch * (128 * EDIM * 2);
        #pragma unroll
        for (int i = 0; i < 16; ++i) {
            unsigned off16 = (unsigned)(i * 256 + tid) * 16u;  // 16B per lane
            unsigned lofs  = sE_off + off16;
            unsigned long long g = (unsigned long long)(srcBf + off16);
            asm volatile("global_load_async_to_lds_b128 %0, %1, off"
                         :: "v"(lofs), "v"(g) : "memory");
        }
        if (tid < 128) sE2[tid] = e2[ch * 128 + tid];
        asm volatile("s_wait_asynccnt 0" ::: "memory");
        __syncthreads();

        if (ch < 7)
            __builtin_prefetch((const char*)ebf + (ch + 1) * (128 * EDIM * 2)
                               + tid * 256, 0, 0);

        #pragma unroll
        for (int t = 0; t < 8; ++t) {
            const int codeTile = ch * 128 + t * 16;
            v8f acc = {};
            #pragma unroll
            for (int f = 0; f < 8; ++f) {
                // A(16x32) bf16 layout: row = lane%16; lane<16: K 0-7 & 16-23,
                // lane>=16: K 8-15 & 24-31 -> two contiguous 16B LDS reads.
                const bf16_t* p = &sE[(t * 16 + col) * EDIM + f * 32 + half * 8];
                v8bf lo = *(const v8bf*)p;
                v8bf hi = *(const v8bf*)(p + 16);
                v16bf av;
                #pragma unroll
                for (int e = 0; e < 8; ++e) { av[e] = lo[e]; av[e + 8] = hi[e]; }
                acc = __builtin_amdgcn_wmma_f32_16x16x32_bf16(
                          false, av, false, B[f], (short)0, acc, false, false);
            }
            // D layout: VGPR j -> code row j (lanes<16) / j+8 (lanes>=16)
            #pragma unroll
            for (int j = 0; j < 8; ++j) {
                int   code = codeTile + half * 8 + j;
                float d    = sE2[t * 16 + half * 8 + j] - 2.0f * acc[j];
                if (d < minv) { minv = d; mini = code; }
            }
        }
    }

    // merge the two half-lane code ranges for the same z-vector
    {
        float om = __shfl_xor(minv, 16, 32);
        int   oi = __shfl_xor(mini, 16, 32);
        if (om < minv || (om == minv && oi < mini)) { minv = om; mini = oi; }
    }

    if (lane < 16) {
        oIdx[n] = (float)mini;                  // indices emitted as float
        atomicAdd(&wsCounts[mini], 1.0f);
    }

    // ---- fused epilogue: z_q gather/store, loss, EMA scatter ----
    const float* ev = emb + mini * EDIM;        // old codebook row (fp32)
    float lsum = 0.0f;
    const int subk = lane >> 4;                 // lanes<16: even k, >=16: odd k
    for (int kk = 0; kk < 128; ++kk) {
        int   k  = kk * 2 + subk;
        float e  = ev[k];
        float zv = z[base + k * HW];
        oZq[base + k * HW] = e;                 // z_q_st == z_q numerically
        float df = e - zv;
        lsum += df * df;
        atomicAdd(&wsEmbsum[mini * EDIM + k], zv);
    }
    #pragma unroll
    for (int off = 16; off; off >>= 1) lsum += __shfl_xor(lsum, off, 32);
    if (lane == 0) atomicAdd(wsLoss, lsum);
}

// ---------------- EMA finalize (single block, 1024 threads) ----------------
__global__ __launch_bounds__(1024)
void vq_final(const float* __restrict__ embAvg, const float* __restrict__ cs,
              const float* __restrict__ wsCounts, const float* __restrict__ wsEmbsum,
              const float* __restrict__ wsLoss,
              float* __restrict__ oLoss, float* __restrict__ oEmb,
              float* __restrict__ oAvg,  float* __restrict__ oCs)
{
    __shared__ float sSm[N_E];
    __shared__ float sRed[32];
    __shared__ float sTot;

    const int t = threadIdx.x, lane = t & 31, w = t >> 5;

    float ncs = cs[t] * DECAYF + (1.0f - DECAYF) * wsCounts[t];
    oCs[t] = ncs;

    float s = ncs;
    #pragma unroll
    for (int off = 16; off; off >>= 1) s += __shfl_xor(s, off, 32);
    if (lane == 0) sRed[w] = s;
    __syncthreads();
    if (t == 0) {
        float tot = 0.0f;
        for (int i = 0; i < 32; ++i) tot += sRed[i];
        sTot = tot;
    }
    __syncthreads();
    const float ntot = sTot;
    sSm[t] = (ncs + EPSF) / (ntot + (float)N_E * EPSF) * ntot;
    __syncthreads();

    for (int i = t; i < N_E * EDIM; i += 1024) {
        float navg = embAvg[i] * DECAYF + (1.0f - DECAYF) * wsEmbsum[i];
        oAvg[i] = navg;
        oEmb[i] = navg / sSm[i >> 8];
    }
    if (t == 0) oLoss[0] = 2.0f * wsLoss[0] / (float)(NVEC * EDIM); // BETA=1
}

// ---------------- launcher ----------------
extern "C" void kernel_launch(void* const* d_in, const int* in_sizes, int n_in,
                              void* d_out, int out_size, void* d_ws, size_t ws_size,
                              hipStream_t stream)
{
    const float* z    = (const float*)d_in[0];   // [32,256,64,64]
    const float* emb  = (const float*)d_in[1];   // [1024,256]
    const float* eavg = (const float*)d_in[2];   // [1024,256]
    const float* cs   = (const float*)d_in[3];   // [1024]

    float* out   = (float*)d_out;
    float* oZq   = out;                          // 33554432
    float* oLoss = out + 33554432;               // 1
    float* oIdx  = out + 33554433;               // 131072
    float* oEmb  = out + 33685505;               // 262144
    float* oAvg  = out + 33947649;               // 262144
    float* oCs   = out + 34209793;               // 1024

    float* ws       = (float*)d_ws;
    float* wsCounts = ws;                        // [0, 1024)
    float* wsEmbsum = ws + 1024;                 // [1024, 263168)
    float* wsLoss   = ws + 263168;               // [263168]
    float* wsE2     = ws + 263172;               // [263172, 264196) (16B aligned)
    bf16_t* wsEbf   = (bf16_t*)(ws + 264196);    // 262144 bf16 = 512 KB

    vq_zero <<<1028, 256, 0, stream>>>(ws, 263169);
    vq_cvt  <<<256,  256, 0, stream>>>(emb, wsEbf);
    vq_e2   <<<4,    256, 0, stream>>>(emb, wsE2);
    vq_main <<<1024, 256, 0, stream>>>(z, emb, wsEbf, wsE2, oZq, oIdx,
                                       wsCounts, wsEmbsum, wsLoss);
    vq_final<<<1,   1024, 0, stream>>>(eavg, cs, wsCounts, wsEmbsum, wsLoss,
                                       oLoss, oEmb, oAvg, oCs);
}
